// MoonshotAttention_21440476741867
// MI455X (gfx1250) — matrix operand order repaired and checked by reading, not compiled, not run on hardware
//
#include <hip/hip_runtime.h>
#include <hip/hip_bf16.h>

typedef __attribute__((ext_vector_type(16))) __bf16 v16bf;
typedef __attribute__((ext_vector_type(8)))  __bf16 v8bf;
typedef __attribute__((ext_vector_type(8)))  float  v8f;

#define SEQ    2048
#define HID    4096
#define NH     32
#define NKV    8
#define HD     128

#if defined(__has_builtin)
#if __has_builtin(__builtin_amdgcn_global_load_async_to_lds_b128)
#define HAVE_ASYNC_LDS 1
#endif
#endif

// 16-byte payload type the async-LDS builtin expects (per hipcc diagnostic):
// 'int __attribute__((vector_size(16)))' pointers, global (as1) src, LDS (as3) dst.
typedef int b128_t __attribute__((vector_size(4 * sizeof(int))));
#define ASYNC_SRC(p) ((__attribute__((address_space(1))) b128_t*)(void*)(p))
#define ASYNC_DST(p) ((__attribute__((address_space(3))) b128_t*)(void*)(p))

__device__ __forceinline__ v8f vzero() {
  v8f z = {0.f, 0.f, 0.f, 0.f, 0.f, 0.f, 0.f, 0.f};
  return z;
}

// A-operand fragment (16x32 bf16): lane = {kh,m}; VGPR0-3: K = kh*8..kh*8+7,
// VGPR4-7: K = 16+kh*8.. (16-bit A layout, row-major source, contiguous K)
__device__ __forceinline__ v16bf ld_frag_a(const __bf16* p, int lh) {
  v8bf lo = *(const v8bf*)(p + lh * 8);
  v8bf hi = *(const v8bf*)(p + 16 + lh * 8);
  v16bf r;
#pragma unroll
  for (int i = 0; i < 8; ++i) { r[i] = lo[i]; r[i + 8] = hi[i]; }
  return r;
}

// B-operand fragment (32x16 bf16): lane = {kh,n}; V0-3: K = kh*8..kh*8+7,
// V4-7: K = 16+kh*8.. (same K order as A fragment).
__device__ __forceinline__ v16bf ld_frag_b(const __bf16* p, int lh) {
  v8bf lo = *(const v8bf*)(p + lh * 8);
  v8bf hi = *(const v8bf*)(p + 16 + lh * 8);
  v16bf r;
#pragma unroll
  for (int i = 0; i < 8; ++i) { r[i] = lo[i]; r[i + 8] = hi[i]; }
  return r;
}

// ---------------------------------------------------------------------------
// Tiled GEMM: C[M][N] = A(bf16, MxK rowmajor) * B(f32, KxN rowmajor) + bias
// block tile 128x128, 8 waves, wave tile 32x64, K-slice 32, double-buffered
// LDS (one barrier per slice). A tile moves via async global->LDS DMA
// (ASYNCcnt path); B tile converts fp32->bf16 in flight
// (v_cvt_pk_bf16_f32 + ds_store_2addr_b32). GL2 prefetch 2 slices ahead.
// ---------------------------------------------------------------------------
#define AST 48   // LDS K-stride (bf16 elems), keeps b128 frag reads aligned
#define BST 48

__global__ __launch_bounds__(256) void gemm_bf16(
    const __bf16* __restrict__ A, const float* __restrict__ B,
    const float* __restrict__ bias, float* __restrict__ C,
    int M, int N, int K)
{
  __shared__ __bf16 As[2][128 * AST];
  __shared__ __bf16 Bs[2][128 * BST];
  const int t = threadIdx.x;
  const int bm = blockIdx.y * 128, bn = blockIdx.x * 128;
  const int wave = t >> 5, lane = t & 31, lh = lane >> 4, ln = lane & 15;
  const int wm = (wave & 3) * 32, wn = (wave >> 2) * 64;

  v8f acc[2][4];
#pragma unroll
  for (int i = 0; i < 2; ++i)
#pragma unroll
    for (int j = 0; j < 4; ++j) acc[i][j] = vzero();

  auto load_tile = [&](int k0, int buf) {
    // A tile: 128 rows x 32 k bf16, 512 16B chunks, 2 per thread.
#pragma unroll
    for (int i = 0; i < 2; ++i) {
      int c = t + i * 256;
      int r = c >> 2, off = (c & 3) * 8;
#if HAVE_ASYNC_LDS
      __builtin_amdgcn_global_load_async_to_lds_b128(
          ASYNC_SRC(A + (size_t)(bm + r) * K + k0 + off),
          ASYNC_DST(&As[buf][r * AST + off]), 0, 0);
#else
      *(v8bf*)(&As[buf][r * AST + off]) =
          *(const v8bf*)(A + (size_t)(bm + r) * K + k0 + off);
#endif
    }
    // B tile: 32 k x 128 n fp32 -> bf16, transposed into Bs[n][k].
    // Load two adjacent k-rows per thread, pack pairs -> aligned b32 stores.
#pragma unroll
    for (int i = 0; i < 2; ++i) {
      int idx = t + i * 256;            // 0..511
      int k2 = (idx >> 5) * 2;          // even k: 0,2,..,30
      int n4 = (idx & 31) * 4;
      const float4 a0 = *(const float4*)(B + (size_t)(k0 + k2) * N + bn + n4);
      const float4 a1 = *(const float4*)(B + (size_t)(k0 + k2 + 1) * N + bn + n4);
      float lo[4] = {a0.x, a0.y, a0.z, a0.w};
      float hi[4] = {a1.x, a1.y, a1.z, a1.w};
#pragma unroll
      for (int jj = 0; jj < 4; ++jj) {
        union { __bf16 h[2]; unsigned u; } pk;
        pk.h[0] = (__bf16)lo[jj];
        pk.h[1] = (__bf16)hi[jj];
        *(unsigned*)&Bs[buf][(n4 + jj) * BST + k2] = pk.u;
      }
    }
  };

  const int nslice = K >> 5;
  load_tile(0, 0);
  for (int ks = 0; ks < nslice; ++ks) {
#if HAVE_ASYNC_LDS
    // our async A-tile stores must land before the slice barrier
#if __has_builtin(__builtin_amdgcn_s_wait_asynccnt)
    __builtin_amdgcn_s_wait_asynccnt(0);
#else
    asm volatile("s_wait_asynccnt 0x0" ::: "memory");
#endif
#endif
    __syncthreads();
    const int buf = ks & 1;
    if (ks + 1 < nslice) load_tile((ks + 1) * 32, buf ^ 1);
    if (ks + 2 < nslice) {
      // warm GL2 for the fp32 weight stream two slices ahead
      int kp = (ks + 2) * 32 + (t >> 3);
      __builtin_prefetch(B + (size_t)kp * N + bn + (t & 7) * 16, 0, 1);
    }

    v16bf af[2], bfr[4];
#pragma unroll
    for (int s = 0; s < 2; ++s)
      af[s] = ld_frag_a(&As[buf][(wm + s * 16 + ln) * AST], lh);
#pragma unroll
    for (int s = 0; s < 4; ++s)
      bfr[s] = ld_frag_b(&Bs[buf][(wn + s * 16 + ln) * BST], lh);
#pragma unroll
    for (int i = 0; i < 2; ++i)
#pragma unroll
      for (int j = 0; j < 4; ++j)
        acc[i][j] = __builtin_amdgcn_wmma_f32_16x16x32_bf16(
            false, af[i], false, bfr[j], (short)0, acc[i][j], false, false);
  }

  // C layout: VGPR r -> M = r (lanes 0-15) / r+8 (lanes 16-31); N = lane%16.
#pragma unroll
  for (int i = 0; i < 2; ++i)
#pragma unroll
    for (int j = 0; j < 4; ++j) {
      int col = bn + wn + j * 16 + ln;
      float bia = bias ? bias[col] : 0.f;
      size_t base = (size_t)(bm + wm + i * 16 + lh * 8) * N + col;
#pragma unroll
      for (int r = 0; r < 8; ++r)
        C[base + (size_t)r * N] = acc[i][j][r] + bia;
    }
}

// ---------------------------------------------------------------------------
// RoPE + pack fp32 -> bf16. One thread per (pos, head, dim-pair).
// ---------------------------------------------------------------------------
__global__ void rope_pack(const float* __restrict__ X, __bf16* __restrict__ Y,
                          const int* __restrict__ pos, int nheads)
{
  int idx = blockIdx.x * blockDim.x + threadIdx.x;
  int total = SEQ * nheads * 64;
  if (idx >= total) return;
  int d = idx & 63;
  int rest = idx >> 6;
  int h = rest % nheads;
  int p = rest / nheads;
  size_t base = ((size_t)p * nheads + h) * HD;
  float x1 = X[base + d], x2 = X[base + d + 64];
  float pf = (float)pos[p];
  // inv_freq = 10000^(-2d/128) = exp(-(2d/128) * ln(10000))
  float inv = __expf(-(float)(2 * d) * (1.0f / 128.0f) * 9.210340371976184f);
  float ang = pf * inv;
  float s, c;
  __sincosf(ang, &s, &c);
  Y[base + d]      = (__bf16)(x1 * c - x2 * s);
  Y[base + d + 64] = (__bf16)(x2 * c + x1 * s);
}

__global__ void cvt_bf16(const float* __restrict__ X, __bf16* __restrict__ Y, int n)
{
  int i = blockIdx.x * blockDim.x + threadIdx.x;
  if (i < n) Y[i] = (__bf16)X[i];
}

// V fp32 [seq][kvh][hd] -> V^T bf16 [kvh][hd][seq] (B-operand ready: per
// output column d, K = seq positions are contiguous).
__global__ void v_transpose(const float* __restrict__ V, __bf16* __restrict__ Vt)
{
  int idx = blockIdx.x * blockDim.x + threadIdx.x;
  if (idx >= SEQ * NKV * HD) return;
  int d = idx % HD;
  int h = (idx / HD) % NKV;
  int p = idx / (HD * NKV);
  Vt[((size_t)h * HD + d) * SEQ + p] = (__bf16)V[idx];
}

// ---------------------------------------------------------------------------
// Flash attention (causal, GQA 4:1). grid = (SEQ/128, NH), 8 waves/WG.
// Each wave owns 16 query rows of one head, iterates 32-key blocks.
// ---------------------------------------------------------------------------
__global__ __launch_bounds__(256) void attn_kernel(
    const __bf16* __restrict__ Qb, const __bf16* __restrict__ Kb,
    const __bf16* __restrict__ Vt, __bf16* __restrict__ Ao)
{
  __shared__ __bf16 P[8][16 * 48];   // per-wave probs staging (A re-layout)
  const int t = threadIdx.x, wave = t >> 5, lane = t & 31;
  const int lh = lane >> 4, ln = lane & 15;
  const int h = blockIdx.y, kvh = h >> 2;
  const int q0 = blockIdx.x * 128 + wave * 16;
  const float scale = 0.08838834764831845f;  // 1/sqrt(128)
  __bf16* Pw = &P[wave][0];

  // Q A-fragments, 16 rows x 128 dim = 4 K-chunks, held in registers.
  v16bf qf[4];
  {
    const __bf16* qp = Qb + (size_t)(q0 + ln) * HID + h * HD;
#pragma unroll
    for (int c = 0; c < 4; ++c) qf[c] = ld_frag_a(qp + c * 32, lh);
  }

  v8f o[8];
  float mr[8], lr[8];
#pragma unroll
  for (int r = 0; r < 8; ++r) { o[r] = vzero(); mr[r] = -1e30f; lr[r] = 0.f; }

  const int jmax = (q0 + 15) >> 5;
  for (int j = 0; j <= jmax; ++j) {
    const int kb = j * 32;
    // S = Q K^T for 32 keys: two 16x16 f32 accumulators.
    v8f s0 = vzero(), s1 = vzero();
    {
      const __bf16* kp0 = Kb + (size_t)(kb + ln) * (NKV * HD) + kvh * HD;
      const __bf16* kp1 = kp0 + (size_t)16 * (NKV * HD);
#pragma unroll
      for (int c = 0; c < 4; ++c) {
        v16bf b0 = ld_frag_b(kp0 + c * 32, lh);
        v16bf b1 = ld_frag_b(kp1 + c * 32, lh);
        s0 = __builtin_amdgcn_wmma_f32_16x16x32_bf16(false, qf[c], false, b0,
                                                     (short)0, s0, false, false);
        s1 = __builtin_amdgcn_wmma_f32_16x16x32_bf16(false, qf[c], false, b1,
                                                     (short)0, s1, false, false);
      }
    }
    // Online softmax. Row m = r + 8*lh lives on the 16 lanes of this half.
#pragma unroll
    for (int r = 0; r < 8; ++r) {
      const int qrow = q0 + r + lh * 8;
      float v0 = s0[r] * scale; if (kb + ln > qrow)      v0 = -1e30f;
      float v1 = s1[r] * scale; if (kb + 16 + ln > qrow) v1 = -1e30f;
      float tm = fmaxf(v0, v1);
#pragma unroll
      for (int msk = 1; msk < 16; msk <<= 1)
        tm = fmaxf(tm, __shfl_xor(tm, msk, 32));
      const float nm = fmaxf(mr[r], tm);
      const float al = __expf(mr[r] - nm);
      v0 = __expf(v0 - nm);
      v1 = __expf(v1 - nm);
      float rs = v0 + v1;
#pragma unroll
      for (int msk = 1; msk < 16; msk <<= 1)
        rs += __shfl_xor(rs, msk, 32);
      lr[r] = al * lr[r] + rs;
      mr[r] = nm;
#pragma unroll
      for (int tt = 0; tt < 8; ++tt) o[tt][r] *= al;
      __bf16* pp = &Pw[(r + lh * 8) * 48];
      pp[ln]      = (__bf16)v0;
      pp[16 + ln] = (__bf16)v1;
    }
    asm volatile("s_wait_dscnt 0x0" ::: "memory");   // LDS write -> read, same wave
    // P (16x32) back as an A-operand, then P·V into 8 output tiles.
    v16bf pf = ld_frag_a(&Pw[ln * 48], lh);
#pragma unroll
    for (int tt = 0; tt < 8; ++tt) {
      const __bf16* vp = Vt + ((size_t)kvh * HD + tt * 16 + ln) * SEQ + kb;
      v16bf bv = ld_frag_b(vp, lh);
      o[tt] = __builtin_amdgcn_wmma_f32_16x16x32_bf16(false, pf, false, bv,
                                                      (short)0, o[tt], false, false);
    }
  }

#pragma unroll
  for (int r = 0; r < 8; ++r) {
    const float inv = 1.0f / lr[r];
    const int row = q0 + r + lh * 8;
#pragma unroll
    for (int tt = 0; tt < 8; ++tt)
      Ao[(size_t)row * HID + h * HD + tt * 16 + ln] = (__bf16)(o[tt][r] * inv);
  }
}

// ---------------------------------------------------------------------------
extern "C" void kernel_launch(void* const* d_in, const int* in_sizes, int n_in,
                              void* d_out, int out_size, void* d_ws, size_t ws_size,
                              hipStream_t stream) {
  (void)in_sizes; (void)n_in; (void)out_size; (void)ws_size;
  const float* X   = (const float*)d_in[0];
  const int*   pos = (const int*)d_in[1];
  const float* Wq  = (const float*)d_in[2];
  const float* bq  = (const float*)d_in[3];
  const float* Wk  = (const float*)d_in[4];
  const float* bk  = (const float*)d_in[5];
  const float* Wv  = (const float*)d_in[6];
  const float* bvv = (const float*)d_in[7];
  const float* Wo  = (const float*)d_in[8];
  float* out = (float*)d_out;

  char* ws = (char*)d_ws;
  __bf16* Xb = (__bf16*)(ws);                      // 16 MB  X in bf16
  float*  Qf = (float*) (ws + (16ull << 20));      // 32 MB  Q fp32
  float*  Kf = (float*) (ws + (48ull << 20));      //  8 MB  K fp32
  float*  Vf = (float*) (ws + (56ull << 20));      //  8 MB  V fp32
  __bf16* Qb = (__bf16*)(ws + (64ull << 20));      // 16 MB  Q rope bf16
  __bf16* Kb = (__bf16*)(ws + (80ull << 20));      //  4 MB  K rope bf16
  __bf16* Vt = (__bf16*)(ws + (84ull << 20));      //  4 MB  V^T bf16
  __bf16* Ao = (__bf16*)(ws + (88ull << 20));      // 16 MB  attn out bf16
  // total 104 MB of d_ws

  { int n = SEQ * HID;
    cvt_bf16<<<(n + 255) / 256, 256, 0, stream>>>(X, Xb, n); }

  gemm_bf16<<<dim3(HID / 128, SEQ / 128), 256, 0, stream>>>(Xb, Wq, bq, Qf, SEQ, HID, HID);
  gemm_bf16<<<dim3((NKV * HD) / 128, SEQ / 128), 256, 0, stream>>>(Xb, Wk, bk, Kf, SEQ, NKV * HD, HID);
  gemm_bf16<<<dim3((NKV * HD) / 128, SEQ / 128), 256, 0, stream>>>(Xb, Wv, bvv, Vf, SEQ, NKV * HD, HID);

  { int n = SEQ * NH * 64;
    rope_pack<<<(n + 255) / 256, 256, 0, stream>>>(Qf, Qb, pos, NH); }
  { int n = SEQ * NKV * 64;
    rope_pack<<<(n + 255) / 256, 256, 0, stream>>>(Kf, Kb, pos, NKV); }
  { int n = SEQ * NKV * HD;
    v_transpose<<<(n + 255) / 256, 256, 0, stream>>>(Vf, Vt); }

  attn_kernel<<<dim3(SEQ / 128, NH), 256, 0, stream>>>(Qb, Kb, Vt, Ao);

  gemm_bf16<<<dim3(HID / 128, SEQ / 128), 256, 0, stream>>>(Ao, Wo, nullptr, out, SEQ, HID, HID);
}
